// FasterRCNN_86277303042048
// MI455X (gfx1250) — matrix operand order repaired and checked by reading, not compiled, not run on hardware
//
#include <hip/hip_runtime.h>
#include <hip/hip_bf16.h>
#include <math.h>

// ---------------------------------------------------------------------------
// Faster R-CNN forward for gfx1250 (MI455X), wave32 + WMMA f16 (f32 accum).
// Conv:  implicit GEMM; each wave computes 4 N-tiles reusing one im2col A
//        fragment (scattered loads amortized 4x).
// FC:    M=64 GEMM; each wave carries all 4 M-tiles so the (huge) weight
//        matrix streams exactly once; A chunk staged into LDS with
//        global_load_async_to_lds_b128 (+ s_wait_asynccnt) per CDNA5 ISA ch8.
// ---------------------------------------------------------------------------

typedef __attribute__((ext_vector_type(16))) _Float16 v16h;
typedef __attribute__((ext_vector_type(8)))  float    v8f;

// WMMA 16x16x32 f16 operand layouts (wave32) as used here (contiguous k, same order for A and B):
// A (16x32): lane L holds row M=L%16; element e -> K = (L>=16?16:0) + e
// B (32x16): lane L holds col N=L%16; element e -> K = (L>=16?16:0) + e
// C/D (16x16 f32): lane L col N=L%16; VGPR r -> M = r + (L>=16?8:0)
__device__ __forceinline__ int kA_of(int lane, int e) {
  return ((lane & 16) ? 16 : 0) + e;
}
__device__ __forceinline__ int kB_of(int lane, int e) {
  return ((lane & 16) ? 16 : 0) + e;
}

// ------------------------------ f32 -> f16 ---------------------------------
__global__ void cvt_f32_f16(const float* __restrict__ in, _Float16* __restrict__ out, int n) {
  int i = blockIdx.x * blockDim.x + threadIdx.x;
  if (i < n) out[i] = (_Float16)in[i];
}

// --------------------- implicit-GEMM conv (1x1 or 3x3) ---------------------
// in : f16 [Cin,H,W], w: f32 OIHW [Cout,Cin,ks,ks], bias f32 [Cout]
// out_h: f16 [Cout,H,W] (optional), out_f: f32 [Cout,H,W] (optional)
// Each wave: 16 pixels x 64 output channels (4 WMMA tiles sharing one A frag).
// Requires H*W % 16 == 0 (true for all layers here).
__global__ void conv_wmma(const _Float16* __restrict__ in, const float* __restrict__ w,
                          const float* __restrict__ bias,
                          _Float16* __restrict__ out_h, float* __restrict__ out_f,
                          int H, int W, int Cin, int Cout, int ks, int relu) {
  const int lane = threadIdx.x;                              // 0..31
  const int mt   = blockIdx.x;                               // pixel tile
  const int g    = blockIdx.y * blockDim.y + threadIdx.y;    // group of 4 n-tiles
  const int HW   = H * W;
  const int ntiles16 = (Cout + 15) >> 4;
  const int ngroups  = (ntiles16 + 3) >> 2;
  if (g >= ngroups) return;                                  // wave-uniform, no barriers
  const int Ktot   = Cin * ks * ks;
  const int nlane0 = g * 64 + (lane & 15);
  const int p      = mt * 16 + (lane & 15);                  // output pixel
  const int py = p / W, px = p % W;

  v8f acc[4] = {};
  for (int kb = 0; kb < Ktot; kb += 32) {
    // --- im2col A fragment (shared across the 4 N-tiles) ---
    v16h a;
#pragma unroll
    for (int e = 0; e < 16; ++e) {
      int k = kb + kA_of(lane, e);
      _Float16 av = (_Float16)0.0f;
      if (k < Ktot) {
        int cin, iy, ix;
        if (ks == 3) {
          cin = k / 9; int r = k - cin * 9;
          iy = py + (r / 3) - 1; ix = px + (r % 3) - 1;
        } else { cin = k; iy = py; ix = px; }
        if (iy >= 0 && iy < H && ix >= 0 && ix < W)
          av = in[cin * HW + iy * W + ix];
      }
      a[e] = av;
    }
    // --- 4 weight fragments, 4 WMMA issues ---
#pragma unroll
    for (int j = 0; j < 4; ++j) {
      const int n = nlane0 + j * 16;
      v16h b;
#pragma unroll
      for (int e = 0; e < 16; ++e) {
        int kk = kb + kB_of(lane, e);
        float bv = (kk < Ktot && n < Cout) ? w[n * Ktot + kk] : 0.0f;
        b[e] = (_Float16)bv;
      }
      if (j == 0) __builtin_prefetch(&w[n * Ktot + kb + 32], 0, 1);  // global_prefetch_b8
      acc[j] = __builtin_amdgcn_wmma_f32_16x16x32_f16(false, a, false, b,
                                                      (short)0, acc[j], false, false);
    }
  }

#pragma unroll
  for (int j = 0; j < 4; ++j) {
    const int n = nlane0 + j * 16;
    if (n < Cout) {
      float bi = bias[n];
#pragma unroll
      for (int r = 0; r < 8; ++r) {
        int mm = mt * 16 + r + ((lane & 16) ? 8 : 0);
        float v = acc[j][r] + bi;
        if (relu) v = fmaxf(v, 0.0f);
        if (out_h) out_h[n * HW + mm] = (_Float16)v;
        if (out_f) out_f[n * HW + mm] = v;
      }
    }
  }
}

// ------------------------------ maxpool 2x2 --------------------------------
__global__ void maxpool2(const _Float16* __restrict__ in, _Float16* __restrict__ out,
                         int C, int H, int W) {
  int Ho = H >> 1, Wo = W >> 1;
  int idx = blockIdx.x * blockDim.x + threadIdx.x;
  if (idx >= C * Ho * Wo) return;
  int c = idx / (Ho * Wo), rem = idx % (Ho * Wo);
  int y = rem / Wo, x = rem % Wo;
  const _Float16* p = in + c * H * W + (2 * y) * W + 2 * x;
  float m = fmaxf(fmaxf((float)p[0], (float)p[1]),
                  fmaxf((float)p[W], (float)p[W + 1]));
  out[idx] = (_Float16)m;
}

// ------------------- proposal generation + top-512 + NMS -------------------
#define NA   3600      // 20*20*9 anchors
#define NTH  256
__global__ void proposals_kernel(const float* __restrict__ score, const float* __restrict__ loc,
                                 float* __restrict__ sc, float* __restrict__ bxw,
                                 float* __restrict__ rois, float* __restrict__ prop_out) {
  const int tid = threadIdx.x;
  // Phase 1: fg score (softmax pair -> sigmoid(s1-s0)), box decode, validity
  for (int i = tid; i < NA; i += NTH) {
    int a = i % 9, pix = i / 9;
    int x = pix % 20, y = pix / 20;
    float s0 = score[(2 * a) * 400 + pix];
    float s1 = score[(2 * a + 1) * 400 + pix];
    float fg = 1.0f / (1.0f + __expf(s0 - s1));
    float l0 = loc[(4 * a + 0) * 400 + pix];
    float l1 = loc[(4 * a + 1) * 400 + pix];
    float l2 = loc[(4 * a + 2) * 400 + pix];
    float l3 = loc[(4 * a + 3) * 400 + pix];
    const float ratios[3] = {0.5f, 1.0f, 2.0f};
    const float scales[3] = {8.0f, 16.0f, 32.0f};
    float r = ratios[a / 3], s = scales[a % 3];
    float h0 = 16.0f * s * sqrtf(r), w0 = 16.0f * s * sqrtf(1.0f / r);
    float cx = x * 16.0f + 8.0f + w0 * l0;
    float cy = y * 16.0f + 8.0f + h0 * l1;
    const float LOG_CLIP = 4.135166556742356f;
    float ww = w0 * __expf(fminf(l2, LOG_CLIP));
    float hh = h0 * __expf(fminf(l3, LOG_CLIP));
    float x1 = fminf(fmaxf(cx - ww * 0.5f, 0.0f), 320.0f);
    float y1 = fminf(fmaxf(cy - hh * 0.5f, 0.0f), 320.0f);
    float x2 = fminf(fmaxf(cx + ww * 0.5f, 0.0f), 320.0f);
    float y2 = fminf(fmaxf(cy + hh * 0.5f, 0.0f), 320.0f);
    bxw[i * 4 + 0] = x1; bxw[i * 4 + 1] = y1;
    bxw[i * 4 + 2] = x2; bxw[i * 4 + 3] = y2;
    bool valid = (x2 - x1 >= 16.0f) && (y2 - y1 >= 16.0f);
    sc[i] = valid ? fg : -__builtin_inff();
  }
  __syncthreads();
  // Phase 2: exact PRE=512 rank cutoff (O(N^2), N=3600 is tiny)
  bool drop[15];
  {
    int t = 0;
    for (int i = tid; i < NA; i += NTH, ++t) {
      float vi = sc[i]; int cnt = 0;
      for (int j = 0; j < NA; ++j) cnt += (sc[j] > vi) ? 1 : 0;
      drop[t] = (cnt >= 512);
    }
    __syncthreads();
    t = 0;
    for (int i = tid; i < NA; i += NTH, ++t)
      if (drop[t]) sc[i] = -__builtin_inff();
    __syncthreads();
  }
  // Phase 3: 64-iteration NMS (argmax reduce + IoU suppression)
  __shared__ float rv[NTH];
  __shared__ int   ri[NTH];
  __shared__ float sbox[4];
  for (int it = 0; it < 64; ++it) {
    float bv = -__builtin_inff(); int bi = 0;
    for (int i = tid; i < NA; i += NTH) {
      float v = sc[i];
      if (v > bv) { bv = v; bi = i; }
    }
    rv[tid] = bv; ri[tid] = bi;
    __syncthreads();
    for (int sft = NTH / 2; sft > 0; sft >>= 1) {
      if (tid < sft && rv[tid + sft] > rv[tid]) {
        rv[tid] = rv[tid + sft]; ri[tid] = ri[tid + sft];
      }
      __syncthreads();
    }
    if (tid == 0) {
      int b = ri[0];
      sbox[0] = bxw[b * 4 + 0]; sbox[1] = bxw[b * 4 + 1];
      sbox[2] = bxw[b * 4 + 2]; sbox[3] = bxw[b * 4 + 3];
      rois[it * 4 + 0] = sbox[0]; rois[it * 4 + 1] = sbox[1];
      rois[it * 4 + 2] = sbox[2]; rois[it * 4 + 3] = sbox[3];
      prop_out[it * 5 + 0] = 0.0f;
      prop_out[it * 5 + 1] = sbox[0]; prop_out[it * 5 + 2] = sbox[1];
      prop_out[it * 5 + 3] = sbox[2]; prop_out[it * 5 + 4] = sbox[3];
    }
    __syncthreads();
    float bx1 = sbox[0], by1 = sbox[1], bx2 = sbox[2], by2 = sbox[3];
    float a1 = (bx2 - bx1) * (by2 - by1);
    for (int i = tid; i < NA; i += NTH) {
      float x1 = fmaxf(bx1, bxw[i * 4 + 0]);
      float y1 = fmaxf(by1, bxw[i * 4 + 1]);
      float x2 = fminf(bx2, bxw[i * 4 + 2]);
      float y2 = fminf(by2, bxw[i * 4 + 3]);
      float inter = fmaxf(x2 - x1, 0.0f) * fmaxf(y2 - y1, 0.0f);
      float a2 = (bxw[i * 4 + 2] - bxw[i * 4 + 0]) * (bxw[i * 4 + 3] - bxw[i * 4 + 1]);
      float iou = inter / (a1 + a2 - inter + 1e-9f);
      if (iou >= 0.7f) sc[i] = -__builtin_inff();
    }
    __syncthreads();
  }
}

// ------------------------------- ROI max-pool ------------------------------
// feat f16 [512,20,20]; one block per (channel, roi); thread = 7x7 bin.
__global__ void roi_pool_kernel(const _Float16* __restrict__ feat,
                                const float* __restrict__ rois,
                                float* __restrict__ pooled) {
  int c = blockIdx.x, roi = blockIdx.y;
  int t = threadIdx.x;
  if (t >= 49) return;
  int i = t / 7, j = t % 7;
  float x1 = rintf(rois[roi * 4 + 0] * 0.0625f);   // round-half-even like jnp.round
  float y1 = rintf(rois[roi * 4 + 1] * 0.0625f);
  float x2 = rintf(rois[roi * 4 + 2] * 0.0625f);
  float y2 = rintf(rois[roi * 4 + 3] * 0.0625f);
  float rw = fmaxf(x2 - x1 + 1.0f, 1.0f);
  float rh = fmaxf(y2 - y1 + 1.0f, 1.0f);
  float m = -__builtin_inff();
#pragma unroll
  for (int gy = 0; gy < 4; ++gy) {
    float fy = (i + (gy + 0.5f) * 0.25f) * (1.0f / 7.0f);
    int yy = (int)fminf(fmaxf(floorf(y1 + fy * rh), 0.0f), 19.0f);
#pragma unroll
    for (int gx = 0; gx < 4; ++gx) {
      float fx = (j + (gx + 0.5f) * 0.25f) * (1.0f / 7.0f);
      int xx = (int)fminf(fmaxf(floorf(x1 + fx * rw), 0.0f), 19.0f);
      m = fmaxf(m, (float)feat[c * 400 + yy * 20 + xx]);
    }
  }
  pooled[roi * 25088 + c * 49 + i * 7 + j] = m;
}

// ------------------------------ FC GEMM (M=64) -----------------------------
// C[64,N] = relu?(A[64,K] @ B[K,N] + bias). A,B f32 -> f16 in-register.
// Block = 4 waves; wave w owns n-tile (blockIdx.x*4+w) and ALL 4 m-tiles so
// the streamed weight matrix is read exactly once. A chunk (64x32 f32, 8 KB)
// is staged into LDS with async-to-LDS DMA each K-step.
__global__ void gemm_wmma(const float* __restrict__ A, const float* __restrict__ B,
                          const float* __restrict__ bias, float* __restrict__ C,
                          int N, int K, int relu) {
  __shared__ float Atile[64 * 32];                   // 8 KB
  const int lane = threadIdx.x;
  const int tid  = threadIdx.y * 32 + lane;          // 0..127
  const int nt   = blockIdx.x * 4 + threadIdx.y;     // n-tile for this wave
  const int n    = nt * 16 + (lane & 15);            // may be >= N (guarded)
  // Flat address of a __shared__ object: low 32 bits are the LDS byte address
  // (CDNA5 aperture mapping).
  const unsigned ldsBase = (unsigned)(size_t)(void*)Atile;

  v8f acc[4] = {};
  for (int kb = 0; kb < K; kb += 32) {               // K % 32 == 0 for all layers
    __syncthreads();                                 // previous chunk fully consumed
    // --- async DMA: A[0:64, kb:kb+32] -> LDS (512 x b128) ---
#pragma unroll
    for (int it = 0; it < 4; ++it) {
      int slot = tid + it * 128;                     // 0..511
      int mrow = slot >> 3;                          // 8 x b128 per 32-float row
      int col  = (slot & 7) << 2;                    // float offset within row
      unsigned laddr = ldsBase + (unsigned)slot * 16u;
      unsigned long long gaddr =
          (unsigned long long)(size_t)(A + (size_t)mrow * K + kb + col);
      asm volatile("global_load_async_to_lds_b128 %0, %1, off"
                   :: "v"(laddr), "v"(gaddr) : "memory");
    }
    asm volatile("s_wait_asynccnt 0" ::: "memory");  // ASYNCcnt == 0
    __syncthreads();

    // --- B fragment: streamed weights (f32 -> f16), shared by all 4 m-tiles ---
    v16h b;
#pragma unroll
    for (int e = 0; e < 16; ++e) {
      int kk = kb + kB_of(lane, e);
      b[e] = (_Float16)((n < N) ? B[(size_t)kk * N + n] : 0.0f);
    }
    if (n < N) __builtin_prefetch(&B[(size_t)(kb + 32) * N + n], 0, 1);

    // --- 4 m-tiles: gather A fragments from LDS, 4 WMMA issues ---
#pragma unroll
    for (int mt = 0; mt < 4; ++mt) {
      v16h a;
#pragma unroll
      for (int e = 0; e < 16; ++e) {
        int mloc = mt * 16 + (lane & 15);
        a[e] = (_Float16)Atile[mloc * 32 + kA_of(lane, e)];
      }
      acc[mt] = __builtin_amdgcn_wmma_f32_16x16x32_f16(false, a, false, b,
                                                       (short)0, acc[mt], false, false);
    }
  }

  if (n < N) {
    float bi = bias[n];
#pragma unroll
    for (int mt = 0; mt < 4; ++mt) {
#pragma unroll
      for (int r = 0; r < 8; ++r) {
        int mm = mt * 16 + r + ((lane & 16) ? 8 : 0);
        float v = acc[mt][r] + bi;
        if (relu) v = fmaxf(v, 0.0f);
        C[(size_t)mm * N + n] = v;
      }
    }
  }
}

// ---------------------------------------------------------------------------
extern "C" void kernel_launch(void* const* d_in, const int* in_sizes, int n_in,
                              void* d_out, int out_size, void* d_ws, size_t ws_size,
                              hipStream_t stream) {
  (void)in_sizes; (void)n_in; (void)out_size; (void)ws_size;
  // Input order: x, conv_w[13], conv_b[13], rpn_w, rpn_b, score_w, score_b,
  //              loc_w, loc_b, cls_w[3], cls_b[3], reg_w[3], reg_b[3]
  const float* x = (const float*)d_in[0];
  const float* conv_w[13]; const float* conv_b[13];
  for (int i = 0; i < 13; ++i) {
    conv_w[i] = (const float*)d_in[1 + i];
    conv_b[i] = (const float*)d_in[14 + i];
  }
  const float* rpn_w   = (const float*)d_in[27];
  const float* rpn_b   = (const float*)d_in[28];
  const float* score_w = (const float*)d_in[29];
  const float* score_b = (const float*)d_in[30];
  const float* loc_w   = (const float*)d_in[31];
  const float* loc_b   = (const float*)d_in[32];
  const float* cls_w[3] = {(const float*)d_in[33], (const float*)d_in[34], (const float*)d_in[35]};
  const float* cls_b[3] = {(const float*)d_in[36], (const float*)d_in[37], (const float*)d_in[38]};
  const float* reg_w[3] = {(const float*)d_in[39], (const float*)d_in[40], (const float*)d_in[41]};
  const float* reg_b[3] = {(const float*)d_in[42], (const float*)d_in[43], (const float*)d_in[44]};
  float* out = (float*)d_out;

  // Workspace carve-out (256B aligned)
  char* ws = (char*)d_ws; size_t off = 0;
  auto alloc = [&](size_t bytes) -> void* {
    void* p = (void*)(ws + off);
    off = (off + bytes + 255) & ~(size_t)255;
    return p;
  };
  _Float16* actA  = (_Float16*)alloc((size_t)64 * 320 * 320 * 2);  // ping
  _Float16* actB  = (_Float16*)alloc((size_t)64 * 320 * 320 * 2);  // pong
  _Float16* feat  = (_Float16*)alloc((size_t)512 * 400 * 2);
  _Float16* rpnh  = (_Float16*)alloc((size_t)512 * 400 * 2);
  float* score    = (float*)alloc((size_t)18 * 400 * 4);
  float* loc      = (float*)alloc((size_t)36 * 400 * 4);
  float* sc       = (float*)alloc((size_t)NA * 4);
  float* bxw      = (float*)alloc((size_t)NA * 4 * 4);
  float* rois     = (float*)alloc((size_t)64 * 4 * 4);
  float* pooled   = (float*)alloc((size_t)64 * 25088 * 4);
  float* h1       = (float*)alloc((size_t)64 * 4096 * 4);
  float* h2       = (float*)alloc((size_t)64 * 4096 * 4);

  // --- input f32 -> f16 ---
  {
    int n = 3 * 320 * 320;
    cvt_f32_f16<<<(n + 255) / 256, 256, 0, stream>>>(x, actA, n);
  }

  // --- VGG backbone ---
  const int  couts[13]    = {64, 64, 128, 128, 256, 256, 256, 512, 512, 512, 512, 512, 512};
  const bool poolAfter[13]= {false, true, false, true, false, false, true,
                             false, false, true, false, false, false};
  int H = 320, W = 320, Cin = 3;
  _Float16* cur = actA; _Float16* nxt = actB;
  auto conv = [&](const _Float16* src, const float* cw, const float* cb,
                  _Float16* dst_h, float* dst_f, int h, int w, int cin, int cout,
                  int ks, int relu) {
    int ngroups = (((cout + 15) >> 4) + 3) >> 2;
    dim3 grid(h * w / 16, (ngroups + 3) / 4);
    dim3 blk(32, 4);
    conv_wmma<<<grid, blk, 0, stream>>>(src, cw, cb, dst_h, dst_f,
                                        h, w, cin, cout, ks, relu);
  };
  for (int i = 0; i < 13; ++i) {
    int Cout = couts[i];
    _Float16* dst = (i == 12) ? feat : nxt;
    conv(cur, conv_w[i], conv_b[i], dst, nullptr, H, W, Cin, Cout, 3, 1);
    Cin = Cout;
    if (i == 12) break;
    _Float16* t = cur; cur = nxt; nxt = t;          // cur = conv output
    if (poolAfter[i]) {
      int n = Cout * (H / 2) * (W / 2);
      maxpool2<<<(n + 255) / 256, 256, 0, stream>>>(cur, nxt, Cout, H, W);
      H >>= 1; W >>= 1;
      t = cur; cur = nxt; nxt = t;                  // cur = pooled output
    }
  }
  // feat: f16 [512,20,20]

  // --- RPN head ---
  conv(feat, rpn_w, rpn_b, rpnh, nullptr, 20, 20, 512, 512, 3, 1);
  conv(rpnh, score_w, score_b, nullptr, score, 20, 20, 512, 18, 1, 0);
  conv(rpnh, loc_w,   loc_b,   nullptr, loc,   20, 20, 512, 36, 1, 0);

  // --- proposals (writes proposals to d_out[6720..7040)) ---
  proposals_kernel<<<1, NTH, 0, stream>>>(score, loc, sc, bxw, rois, out + 6720);

  // --- ROI max pooling -> pooled [64, 25088] f32 ---
  roi_pool_kernel<<<dim3(512, 64), 64, 0, stream>>>(feat, rois, pooled);

  // --- cls / reg MLPs ---
  auto gemm = [&](const float* A, const float* B, const float* bias, float* C,
                  int N, int K, int relu) {
    dim3 grid((((N + 15) / 16) + 3) / 4);            // 4 n-tiles per block
    dim3 blk(32, 4);
    gemm_wmma<<<grid, blk, 0, stream>>>(A, B, bias, C, N, K, relu);
  };
  gemm(pooled, cls_w[0], cls_b[0], h1, 4096, 25088, 1);
  gemm(h1,     cls_w[1], cls_b[1], h2, 4096, 4096, 1);
  gemm(h2,     cls_w[2], cls_b[2], out, 21, 4096, 0);          // scores [64,21]
  gemm(pooled, reg_w[0], reg_b[0], h1, 4096, 25088, 1);
  gemm(h1,     reg_w[1], reg_b[1], h2, 4096, 4096, 1);
  gemm(h2,     reg_w[2], reg_b[2], out + 1344, 84, 4096, 0);   // deltas [64,84]
}
